// WinCLIP_69922067579263
// MI455X (gfx1250) — compile-verified, matrix operand order checked by reading, not run on hardware
//
#include <hip/hip_runtime.h>
#include <hip/hip_bf16.h>
#include <math.h>

typedef __attribute__((ext_vector_type(16))) __bf16 v16bf;
typedef __attribute__((ext_vector_type(8)))  __bf16 v8bf;
typedef __attribute__((ext_vector_type(4)))  __bf16 v4bf;
typedef __attribute__((ext_vector_type(8)))  float  v8f;
typedef __attribute__((ext_vector_type(4)))  float  v4f;

#define DIM   640
#define BATCH 256
#define HW    225
#define IMG   240
#define TEMP  0.07f
#define WPB   2            // waves per block in the WMMA kernel

__device__ __forceinline__ float waveReduceSum(float v) {
#pragma unroll
  for (int m = 16; m >= 1; m >>= 1) v += __shfl_xor(v, m, 32);
  return v;
}

// ---------------------------------------------------------------------------
// 1 / max(||row||, 1e-12) per row, one wave32 per row
// ---------------------------------------------------------------------------
__global__ void invnorm_kernel(const float* __restrict__ x,
                               float* __restrict__ invn, int rows) {
  int lane = threadIdx.x & 31;
  int row  = blockIdx.x * (blockDim.x >> 5) + (threadIdx.x >> 5);
  if (row >= rows) return;
  const float* xp = x + (size_t)row * DIM;
  float s = 0.f;
  for (int i = lane; i < DIM; i += 32) { float v = xp[i]; s += v * v; }
  s = waveReduceSum(s);
  if (lane == 0) invn[row] = 1.0f / fmaxf(sqrtf(s), 1e-12f);
}

// ---------------------------------------------------------------------------
// normalize + convert reference rows to bf16 (padded rows -> zeros)
// ---------------------------------------------------------------------------
__global__ void cvt_ref_kernel(const float* __restrict__ x,
                               const float* __restrict__ invn,
                               __bf16* __restrict__ out, int rows, int rowsPad) {
  int lane = threadIdx.x & 31;
  int row  = blockIdx.x * (blockDim.x >> 5) + (threadIdx.x >> 5);
  if (row >= rowsPad) return;
  __bf16* op = out + (size_t)row * DIM;
  if (row >= rows) {
    for (int i = lane; i < DIM / 4; i += 32) {
      v4bf z = {};
      *(v4bf*)(op + i * 4) = z;
    }
    return;
  }
  const float* xp = x + (size_t)row * DIM;
  float s = invn[row];
  for (int i = lane; i < DIM / 4; i += 32) {
    v4f v = *(const v4f*)(xp + i * 4);
    v4bf o;
    o[0] = (__bf16)(v[0] * s); o[1] = (__bf16)(v[1] * s);
    o[2] = (__bf16)(v[2] * s); o[3] = (__bf16)(v[3] * s);
    *(v4bf*)(op + i * 4) = o;
  }
}

// ---------------------------------------------------------------------------
// softmax(cos(x,text)/T)[...,1] per row, one wave32 per row
// ---------------------------------------------------------------------------
__global__ void probs_kernel(const float* __restrict__ x,
                             const float* __restrict__ xinv,
                             const float* __restrict__ txt,
                             const float* __restrict__ tinv,
                             float* __restrict__ out, int rows) {
  int lane = threadIdx.x & 31;
  int row  = blockIdx.x * (blockDim.x >> 5) + (threadIdx.x >> 5);
  if (row >= rows) return;
  const float* xp = x + (size_t)row * DIM;
  float d0 = 0.f, d1 = 0.f;
  for (int i = lane; i < DIM; i += 32) {
    float v = xp[i];
    d0 += v * txt[i];
    d1 += v * txt[DIM + i];
  }
  d0 = waveReduceSum(d0);
  d1 = waveReduceSum(d1);
  if (lane == 0) {
    float xi = xinv[row];
    float c0 = d0 * xi * tinv[0];
    float c1 = d1 * xi * tinv[1];
    out[row] = 1.f / (1.f + __expf((c0 - c1) / TEMP));
  }
}

// ---------------------------------------------------------------------------
// visual association score:  out[q] = min_r (1 - cos(q,r)) / 2
// One wave per 16-query tile. Phase 1: normalize+convert the query tile to
// bf16 in LDS (once). Phase 2: loop ref tiles in blocks of 4 with 4 f32
// accumulators; A fragments from LDS (2 x ds_load_b128), B fragments from the
// preconverted bf16 ref table (2 x global b128) -> 4 wmma per k-step.
// A 16x32 bf16 layout: lane<16 -> row=lane,  K chunks {kk+0..7, kk+16..23}
//                      lane>=16 -> row=lane-16, K chunks {kk+8..15, kk+24..31}
// C/D: lanes 0-15 hold M=0..7 (vgpr j), lanes 16-31 hold M=8..15.
// ---------------------------------------------------------------------------
__global__ void __launch_bounds__(WPB * 32)
vas_wmma_kernel(const float* __restrict__ q, const float* __restrict__ qinv,
                const __bf16* __restrict__ rb, float* __restrict__ out,
                int nR, int nRT4) {
  __shared__ __bf16 sA[WPB][16 * DIM];
  const int lane = threadIdx.x & 31;
  const int wl   = threadIdx.x >> 5;
  const int wave = blockIdx.x * WPB + wl;
  const int half = lane >> 4;
  const int idx  = lane & 15;

  // ---- phase 1: query tile -> normalized bf16 in LDS ----
  {
    const float* qb = q + (size_t)wave * 16 * DIM;
    const float* qi = qinv + (size_t)wave * 16;
    __bf16* sp = sA[wl];
    for (int i = lane; i < 16 * (DIM / 4); i += 32) {
      int row = i / (DIM / 4);
      int c4  = i % (DIM / 4);
      float s = qi[row];
      v4f v = *(const v4f*)(qb + (size_t)row * DIM + c4 * 4);
      v4bf o;
      o[0] = (__bf16)(v[0] * s); o[1] = (__bf16)(v[1] * s);
      o[2] = (__bf16)(v[2] * s); o[3] = (__bf16)(v[3] * s);
      *(v4bf*)(sp + row * DIM + c4 * 4) = o;
    }
  }
  __syncthreads();

  float runMin[8];
#pragma unroll
  for (int j = 0; j < 8; ++j) runMin[j] = 1e30f;

  const __bf16* sArow = sA[wl] + idx * DIM;

  for (int rt4 = 0; rt4 < nRT4; ++rt4) {
    v8f acc[4];
#pragma unroll
    for (int t = 0; t < 4; ++t) acc[t] = (v8f){};
    const __bf16* rbase = rb + (size_t)(rt4 * 64 + idx) * DIM;

#pragma unroll 2
    for (int kk = 0; kk < DIM; kk += 32) {
      const int c0 = kk + half * 8;
      v8bf alo = *(const v8bf*)(sArow + c0);
      v8bf ahi = *(const v8bf*)(sArow + c0 + 16);
      v16bf a = __builtin_shufflevector(alo, ahi,
                  0, 1, 2, 3, 4, 5, 6, 7, 8, 9, 10, 11, 12, 13, 14, 15);
#pragma unroll
      for (int t = 0; t < 4; ++t) {
        const __bf16* rp = rbase + (size_t)t * 16 * DIM;
        v8bf blo = *(const v8bf*)(rp + c0);
        v8bf bhi = *(const v8bf*)(rp + c0 + 16);
        v16bf b = __builtin_shufflevector(blo, bhi,
                    0, 1, 2, 3, 4, 5, 6, 7, 8, 9, 10, 11, 12, 13, 14, 15);
        acc[t] = __builtin_amdgcn_wmma_f32_16x16x32_bf16(
            /*neg_a=*/false, a, /*neg_b=*/false, b,
            /*c_mod=*/(short)0, acc[t], /*reuse_a=*/false, /*reuse_b=*/false);
      }
    }

#pragma unroll
    for (int t = 0; t < 4; ++t) {
      const bool rOK = (rt4 * 64 + t * 16 + idx) < nR;
#pragma unroll
      for (int j = 0; j < 8; ++j) {
        float sc = rOK ? (1.f - acc[t][j]) * 0.5f : 1e30f;
        runMin[j] = fminf(runMin[j], sc);
      }
    }
  }

  // min across the 16 lanes of each half-wave (xor 8,4,2,1 stays in-half)
#pragma unroll
  for (int j = 0; j < 8; ++j) {
#pragma unroll
    for (int m = 8; m >= 1; m >>= 1)
      runMin[j] = fminf(runMin[j], __shfl_xor(runMin[j], m, 32));
  }
  // row M = half*8 + j; lane with idx==j in each half writes it
#pragma unroll
  for (int j = 0; j < 8; ++j) {
    if (idx == j) out[wave * 16 + half * 8 + j] = runMin[j];
  }
}

// ---------------------------------------------------------------------------
// harmonic_aggregation: out[b,p] = counts[p] / sum_{windows covering p} 1/ws[b,w]
// ---------------------------------------------------------------------------
__global__ void harmonic_kernel(const float* __restrict__ ws,
                                const int* __restrict__ masks,
                                float* __restrict__ out, int nW, int k2) {
  __shared__ float sw[200];
  __shared__ int   sm[1536];
  const int b = blockIdx.x;
  const int p = threadIdx.x;            // 0..224
  for (int i = threadIdx.x; i < nW; i += blockDim.x)
    sw[i] = ws[(size_t)b * nW + i];
  for (int i = threadIdx.x; i < k2 * nW; i += blockDim.x)
    sm[i] = masks[i];
  __syncthreads();

  float cnt = 0.f, sum = 0.f;
  for (int i = 0; i < k2; ++i)
    for (int w = 0; w < nW; ++w)
      if (sm[i * nW + w] == p) { cnt += 1.f; sum += 1.f / sw[w]; }
  float rr = cnt / sum;
  out[(size_t)b * HW + p] = __builtin_isfinite(rr) ? rr : 0.f;
}

// ---------------------------------------------------------------------------
__global__ void combine_kernel(const float* __restrict__ img0,
                               const float* __restrict__ zsA2,
                               const float* __restrict__ zsA3,
                               const float* __restrict__ vasP,
                               const float* __restrict__ fsA2,
                               const float* __restrict__ fsA3,
                               float* __restrict__ multi,
                               float* __restrict__ few) {
  int t = blockIdx.x * blockDim.x + threadIdx.x;   // 0..B*HW-1 (exact)
  int b = t / HW;
  float z0 = img0[b], z1 = zsA2[t], z2 = zsA3[t];
  float zs = 3.f / (1.f / z0 + 1.f / z1 + 1.f / z2);
  float fw = (vasP[t] + fsA2[t] + fsA3[t]) * (1.f / 3.f);
  few[t]   = fw;
  multi[t] = 0.5f * (zs + fw);
}

// ---------------------------------------------------------------------------
__global__ void imgscore_kernel(const float* __restrict__ img0,
                                const float* __restrict__ few,
                                float* __restrict__ out_scores) {
  __shared__ float red[256];
  const int b = blockIdx.x;
  float m = -1e30f;
  for (int i = threadIdx.x; i < HW; i += blockDim.x)
    m = fmaxf(m, few[(size_t)b * HW + i]);
  red[threadIdx.x] = m;
  __syncthreads();
  for (int s = 128; s >= 1; s >>= 1) {
    if (threadIdx.x < s) red[threadIdx.x] = fmaxf(red[threadIdx.x], red[threadIdx.x + s]);
    __syncthreads();
  }
  if (threadIdx.x == 0) out_scores[b] = 0.5f * (img0[b] + red[0]);
}

// ---------------------------------------------------------------------------
// 15x15 -> 240x240 bilinear (half-pixel centers; clamped borders reproduce
// jax's renormalized triangle weights for upsampling)
// ---------------------------------------------------------------------------
__global__ void resize_kernel(const float* __restrict__ multi,
                              float* __restrict__ outp) {
  int t = blockIdx.x * blockDim.x + threadIdx.x;   // 0..B*IMG*IMG-1 (exact)
  int x = t % IMG;
  int y = (t / IMG) % IMG;
  int b = t / (IMG * IMG);
  float fy = (y + 0.5f) * (1.f / 16.f) - 0.5f;
  float fx = (x + 0.5f) * (1.f / 16.f) - 0.5f;
  int yf = (int)floorf(fy), xf = (int)floorf(fx);
  float wy = fy - yf, wx = fx - xf;
  int y0 = min(max(yf, 0), 14),     y1 = min(max(yf + 1, 0), 14);
  int x0 = min(max(xf, 0), 14),     x1 = min(max(xf + 1, 0), 14);
  const float* mb = multi + (size_t)b * HW;
  float v00 = mb[y0 * 15 + x0], v01 = mb[y0 * 15 + x1];
  float v10 = mb[y1 * 15 + x0], v11 = mb[y1 * 15 + x1];
  float v0 = v00 + wx * (v01 - v00);
  float v1 = v10 + wx * (v11 - v10);
  outp[t] = v0 + wy * (v1 - v0);
}

// ---------------------------------------------------------------------------
extern "C" void kernel_launch(void* const* d_in, const int* in_sizes, int n_in,
                              void* d_out, int out_size, void* d_ws, size_t ws_size,
                              hipStream_t stream) {
  const float* img = (const float*)d_in[0];
  const float* w2  = (const float*)d_in[1];
  const float* w3  = (const float*)d_in[2];
  const float* pt  = (const float*)d_in[3];
  const float* txt = (const float*)d_in[4];
  const float* rpt = (const float*)d_in[5];
  const float* rw2 = (const float*)d_in[6];
  const float* rw3 = (const float*)d_in[7];
  const int*   m2  = (const int*)d_in[8];
  const int*   m3  = (const int*)d_in[9];

  const int N2 = 196, N3 = 169, NP = 225, K = 4;
  const int RQ2 = BATCH * N2, RQ3 = BATCH * N3, RQP = BATCH * NP;
  const int RR2 = K * N2,     RR3 = K * N3,     RRP = K * NP;
  // ref row counts padded to multiples of 64 (4 tiles of 16)
  const int RR2p = (RR2 + 63) & ~63;   // 832
  const int RR3p = (RR3 + 63) & ~63;   // 704
  const int RRPp = (RRP + 63) & ~63;   // 960

  float* wsp = (float*)d_ws;
  size_t off = 0;
  auto alloc = [&](size_t n) { float* p = wsp + off; off += (n + 63) & ~(size_t)63; return p; };
  float* inv_img = alloc(BATCH);
  float* inv_txt = alloc(2);
  float* inv_w2  = alloc(RQ2);
  float* inv_w3  = alloc(RQ3);
  float* inv_pt  = alloc(RQP);
  float* inv_rpt = alloc(RRP);
  float* inv_rw2 = alloc(RR2);
  float* inv_rw3 = alloc(RR3);
  float* img0    = alloc(BATCH);
  float* pw2     = alloc(RQ2);
  float* pw3     = alloc(RQ3);
  float* vasP    = alloc(RQP);
  float* vasW2   = alloc(RQ2);
  float* vasW3   = alloc(RQ3);
  float* zsA2    = alloc(BATCH * HW);
  float* zsA3    = alloc(BATCH * HW);
  float* fsA2    = alloc(BATCH * HW);
  float* fsA3    = alloc(BATCH * HW);
  float* multi   = alloc(BATCH * HW);
  float* few     = alloc(BATCH * HW);
  __bf16* rbP  = (__bf16*)alloc((size_t)RRPp * DIM / 2);
  __bf16* rbW2 = (__bf16*)alloc((size_t)RR2p * DIM / 2);
  __bf16* rbW3 = (__bf16*)alloc((size_t)RR3p * DIM / 2);

  auto wg = [](int rows) { return (rows + 7) / 8; };   // 8 waves / 256-thread block

  // ---- inverse norms ----
  invnorm_kernel<<<wg(BATCH), 256, 0, stream>>>(img, inv_img, BATCH);
  invnorm_kernel<<<1, 64, 0, stream>>>(txt, inv_txt, 2);
  invnorm_kernel<<<wg(RQ2), 256, 0, stream>>>(w2, inv_w2, RQ2);
  invnorm_kernel<<<wg(RQ3), 256, 0, stream>>>(w3, inv_w3, RQ3);
  invnorm_kernel<<<wg(RQP), 256, 0, stream>>>(pt, inv_pt, RQP);
  invnorm_kernel<<<wg(RRP), 256, 0, stream>>>(rpt, inv_rpt, RRP);
  invnorm_kernel<<<wg(RR2), 256, 0, stream>>>(rw2, inv_rw2, RR2);
  invnorm_kernel<<<wg(RR3), 256, 0, stream>>>(rw3, inv_rw3, RR3);

  // ---- refs -> normalized bf16 (zero padding rows) ----
  cvt_ref_kernel<<<wg(RRPp), 256, 0, stream>>>(rpt, inv_rpt, rbP,  RRP, RRPp);
  cvt_ref_kernel<<<wg(RR2p), 256, 0, stream>>>(rw2, inv_rw2, rbW2, RR2, RR2p);
  cvt_ref_kernel<<<wg(RR3p), 256, 0, stream>>>(rw3, inv_rw3, rbW3, RR3, RR3p);

  // ---- zero-shot text probabilities ----
  probs_kernel<<<wg(BATCH), 256, 0, stream>>>(img, inv_img, txt, inv_txt, img0, BATCH);
  probs_kernel<<<wg(RQ2), 256, 0, stream>>>(w2, inv_w2, txt, inv_txt, pw2, RQ2);
  probs_kernel<<<wg(RQ3), 256, 0, stream>>>(w3, inv_w3, txt, inv_txt, pw3, RQ3);

  // ---- few-shot visual association (WMMA) ----
  // query row counts are multiples of 16*WPB, grids are exact
  vas_wmma_kernel<<<RQP / (16 * WPB), 32 * WPB, 0, stream>>>(pt, inv_pt, rbP,  vasP,  RRP, RRPp / 64);
  vas_wmma_kernel<<<RQ2 / (16 * WPB), 32 * WPB, 0, stream>>>(w2, inv_w2, rbW2, vasW2, RR2, RR2p / 64);
  vas_wmma_kernel<<<RQ3 / (16 * WPB), 32 * WPB, 0, stream>>>(w3, inv_w3, rbW3, vasW3, RR3, RR3p / 64);

  // ---- harmonic aggregation ----
  harmonic_kernel<<<BATCH, 225, 0, stream>>>(pw2,   m2, zsA2, N2, 4);
  harmonic_kernel<<<BATCH, 225, 0, stream>>>(pw3,   m3, zsA3, N3, 9);
  harmonic_kernel<<<BATCH, 225, 0, stream>>>(vasW2, m2, fsA2, N2, 4);
  harmonic_kernel<<<BATCH, 225, 0, stream>>>(vasW3, m3, fsA3, N3, 9);

  // ---- combine ----
  combine_kernel<<<(BATCH * HW) / 256, 256, 0, stream>>>(img0, zsA2, zsA3,
                                                         vasP, fsA2, fsA3,
                                                         multi, few);

  // ---- outputs ----
  float* outScores = (float*)d_out;
  float* outPixels = outScores + BATCH;
  imgscore_kernel<<<BATCH, 256, 0, stream>>>(img0, few, outScores);
  resize_kernel<<<(BATCH * IMG * IMG) / 256, 256, 0, stream>>>(multi, outPixels);
}
